// MyNet_1683627180032
// MI455X (gfx1250) — compile-verified, matrix operand order checked by reading, not run on hardware
//
#include <hip/hip_runtime.h>
#include <hip/hip_bf16.h>

typedef __attribute__((ext_vector_type(16))) _Float16 v16h;
typedef __attribute__((ext_vector_type(8)))  float    v8f;

#define XD 64
#define ED 16
#define KD (XD + ED)   // 80
#define INNER 512

// ---------------------------------------------------------------------------
// Utility kernels
// ---------------------------------------------------------------------------
__global__ void zero_f32_kernel(float* __restrict__ p, long long n) {
  long long i = (long long)blockIdx.x * blockDim.x + threadIdx.x;
  if (i < n) p[i] = 0.0f;
}

__global__ void copy_f32_kernel(float* __restrict__ dst, const float* __restrict__ src, long long n) {
  long long i = (long long)blockIdx.x * blockDim.x + threadIdx.x;
  if (i < n) dst[i] = src[i];
}

// ---------------------------------------------------------------------------
// Weight pre-pack: convert f32 weights to f16 B-fragments in per-lane order.
// Layout: Wp[frag][tile][lane] (v16h each) -> lane's 32B is contiguous and
// adjacent lanes are adjacent => coalesced b128 loads, no cvt in the GEMM loop.
// B-fragment mapping (16-bit, 16x32): lane L holds N = L&15, half = L>>4;
// VGPR v -> K = (v>>2)*16 + half*8 + (v&3)*2 (+ frag*32).
// ---------------------------------------------------------------------------
__global__ void pack_wout_kernel(const float* __restrict__ Wout,  // [512,64] row-major
                                 v16h* __restrict__ Wp) {
  const int tid  = blockIdx.x * blockDim.x + threadIdx.x;   // 0..1023
  if (tid >= 32 * 32) return;
  const int nt   = tid >> 5;       // column tile 0..31
  const int lane = tid & 31;
  const int half = lane >> 4;
  const int col  = nt * 16 + (lane & 15);
  const float* wr = Wout + (long long)col * XD;
  v16h lo, hi;
#pragma unroll
  for (int v = 0; v < 8; ++v) {
    int k = ((v >> 2) << 4) + (half << 3) + ((v & 3) << 1);
    lo[2 * v]     = (_Float16)wr[k];
    lo[2 * v + 1] = (_Float16)wr[k + 1];
    hi[2 * v]     = (_Float16)wr[k + 32];
    hi[2 * v + 1] = (_Float16)wr[k + 33];
  }
  Wp[0 * 1024 + nt * 32 + lane] = lo;
  Wp[1 * 1024 + nt * 32 + lane] = hi;
}

__global__ void pack_win_kernel(const float* __restrict__ Win,    // [64,80] row-major
                                v16h* __restrict__ Wp) {
  const int tid  = blockIdx.x * blockDim.x + threadIdx.x;   // 0..127
  if (tid >= 4 * 32) return;
  const int nt   = tid >> 5;       // column tile 0..3
  const int lane = tid & 31;
  const int half = lane >> 4;
  const int col  = nt * 16 + (lane & 15);
  const float* wr = Win + (long long)col * KD;
#pragma unroll
  for (int s = 0; s < 3; ++s) {
    v16h f;
#pragma unroll
    for (int v = 0; v < 8; ++v) {
      int kk = ((v >> 2) << 4) + (half << 3) + ((v & 3) << 1);
#pragma unroll
      for (int u = 0; u < 2; ++u) {
        int k = s * 32 + kk + u;
        f[2 * v + u] = (_Float16)((k < KD) ? wr[k] : 0.0f);
      }
    }
    Wp[s * 128 + nt * 32 + lane] = f;
  }
}

// ---------------------------------------------------------------------------
// Edge scatter: agg_x[dst] += x[src] (feature-parallel, coalesced per edge)
// ---------------------------------------------------------------------------
__global__ void scatter_x_kernel(const float* __restrict__ x, const int* __restrict__ ei,
                                 float* __restrict__ aggx, int E) {
  long long idx = (long long)blockIdx.x * blockDim.x + threadIdx.x;
  long long total = (long long)E * XD;
  if (idx >= total) return;
  int e = (int)(idx >> 6);      // /64
  int f = (int)(idx & 63);
  int s = ei[e];                // edge_index[0][e]
  int d = ei[E + e];            // edge_index[1][e]
  atomicAdd(&aggx[(long long)d * XD + f], x[(long long)s * XD + f]);
}

__global__ void scatter_e_kernel(const float* __restrict__ ea, const int* __restrict__ ei,
                                 float* __restrict__ agge, int E) {
  long long idx = (long long)blockIdx.x * blockDim.x + threadIdx.x;
  long long total = (long long)E * ED;
  if (idx >= total) return;
  int e = (int)(idx >> 4);      // /16
  int f = (int)(idx & 15);
  int d = ei[E + e];
  atomicAdd(&agge[(long long)d * ED + f], ea[(long long)e * ED + f]);
}

// ---------------------------------------------------------------------------
// Fused: z = x @ W_out^T + b  (WMMA f16->f32), softmax rows in LDS,
// then fp[batch[node]] += softmax(z) via f32 atomics (stays in 192MB L2).
// Block = 256 threads = 8 waves; block owns a 16-node tile; wave w computes
// column tiles w*4 .. w*4+3 of the 512-wide output (2 WMMA k-steps each).
// ---------------------------------------------------------------------------
__global__ __launch_bounds__(256) void softmax_fp_kernel(
    const float* __restrict__ x, const v16h* __restrict__ WoutP,
    const float* __restrict__ bout, const int* __restrict__ batch,
    float* __restrict__ fp, int N) {
  __shared__ float zsm[16][INNER + 8];

  const int base = blockIdx.x * 16;
  const int tid  = threadIdx.x;
  const int lane = tid & 31;
  const int wave = tid >> 5;          // 0..7
  const int half = lane >> 4;         // 0 or 1
  const int nloc = lane & 15;

  // --- A fragments: 16 rows of x, K = 0..63 (two 16x32 f16 frags) ---
  v16h a_lo, a_hi;
  {
    int m = base + nloc;
    if (m >= N) m = N - 1;            // exact tiling when N%16==0; clamp otherwise
    const float* xr = x + (long long)m * XD;
#pragma unroll
    for (int v = 0; v < 8; ++v) {
      int k = ((v >> 2) << 4) + (half << 3) + ((v & 3) << 1);
      a_lo[2 * v]     = (_Float16)xr[k];
      a_lo[2 * v + 1] = (_Float16)xr[k + 1];
      a_hi[2 * v]     = (_Float16)xr[k + 32];
      a_hi[2 * v + 1] = (_Float16)xr[k + 33];
    }
  }

  // --- 4 column tiles per wave ---
#pragma unroll
  for (int t = 0; t < 4; ++t) {
    const int nt  = wave * 4 + t;     // 0..31
    const int col = nt * 16 + nloc;   // 0..511
    const v16h b_lo = WoutP[0 * 1024 + nt * 32 + lane];
    const v16h b_hi = WoutP[1 * 1024 + nt * 32 + lane];
    const float bias = bout[col];
    v8f c = {bias, bias, bias, bias, bias, bias, bias, bias};
    c = __builtin_amdgcn_wmma_f32_16x16x32_f16(false, a_lo, false, b_lo, (short)0, c, false, false);
    c = __builtin_amdgcn_wmma_f32_16x16x32_f16(false, a_hi, false, b_hi, (short)0, c, false, false);
    // C/D layout: VGPR v, lanes 0-15 -> M=v, lanes 16-31 -> M=v+8; N = lane&15
#pragma unroll
    for (int v = 0; v < 8; ++v) {
      int m = v + half * 8;
      zsm[m][col] = c[v];
    }
  }
  __syncthreads();

  // --- softmax per row + scatter into fp (wave w owns rows 2w, 2w+1) ---
#pragma unroll
  for (int rr = 0; rr < 2; ++rr) {
    const int r    = wave * 2 + rr;
    const int node = base + r;
    if (node >= N) continue;
    float mx = -3.4e38f;
    for (int j = lane; j < INNER; j += 32) mx = fmaxf(mx, zsm[r][j]);
#pragma unroll
    for (int o = 16; o > 0; o >>= 1) mx = fmaxf(mx, __shfl_xor(mx, o, 32));
    float sum = 0.0f;
    for (int j = lane; j < INNER; j += 32) {
      float e = __expf(zsm[r][j] - mx);
      zsm[r][j] = e;
      sum += e;
    }
#pragma unroll
    for (int o = 16; o > 0; o >>= 1) sum += __shfl_xor(sum, o, 32);
    const float inv = 1.0f / sum;
    float* dst = fp + (long long)batch[node] * INNER;
    for (int j = lane; j < INNER; j += 32) atomicAdd(dst + j, zsm[r][j] * inv);
  }
}

// ---------------------------------------------------------------------------
// Conv matmul: x_out[N,64] = concat(agg_x[N,64], agg_e[N,16]) @ W_in^T + b
// K = 80, zero-padded to 96 (3 k-steps of 32). Block = 128 threads = 4 waves;
// wave w computes output column tile w of 4.
// ---------------------------------------------------------------------------
__global__ __launch_bounds__(128) void conv_mm_kernel(
    const float* __restrict__ aggx, const float* __restrict__ agge,
    const v16h* __restrict__ WinP, const float* __restrict__ bin,
    float* __restrict__ xout, int N) {
  const int base = blockIdx.x * 16;
  const int lane = threadIdx.x & 31;
  const int wave = threadIdx.x >> 5;   // 0..3
  const int half = lane >> 4;
  const int nloc = lane & 15;

  v16h A[3];
  {
    int m = base + nloc;
    if (m >= N) m = N - 1;
    const float* ax = aggx + (long long)m * XD;
    const float* ae = agge + (long long)m * ED;
#pragma unroll
    for (int s = 0; s < 3; ++s) {
#pragma unroll
      for (int v = 0; v < 8; ++v) {
        int kk = ((v >> 2) << 4) + (half << 3) + ((v & 3) << 1);
#pragma unroll
        for (int u = 0; u < 2; ++u) {
          int k = s * 32 + kk + u;
          float f = (k < XD) ? ax[k] : ((k < KD) ? ae[k - XD] : 0.0f);
          A[s][2 * v + u] = (_Float16)f;
        }
      }
    }
  }
  const int col = wave * 16 + nloc;    // 0..63
  const float bias = bin[col];
  v8f c = {bias, bias, bias, bias, bias, bias, bias, bias};
#pragma unroll
  for (int s = 0; s < 3; ++s) {
    const v16h b = WinP[s * 128 + wave * 32 + lane];
    c = __builtin_amdgcn_wmma_f32_16x16x32_f16(false, A[s], false, b, (short)0, c, false, false);
  }
#pragma unroll
  for (int v = 0; v < 8; ++v) {
    int m = v + half * 8;
    int node = base + m;
    if (node < N) xout[(long long)node * XD + col] = c[v];
  }
}

// ---------------------------------------------------------------------------
// Final MLP: out[g] = sigmoid((fp[g] @ lin1^T + b1) @ lin2^T + b2)
// One wave32 per graph; fp row (512) held lane-strided in 16 registers.
// ---------------------------------------------------------------------------
__global__ __launch_bounds__(256) void mlp_kernel(
    const float* __restrict__ fp, const float* __restrict__ l1w,
    const float* __restrict__ l1b, const float* __restrict__ l2w,
    const float* __restrict__ l2b, float* __restrict__ out, int G) {
  const int gid  = blockIdx.x * 8 + (threadIdx.x >> 5);
  const int lane = threadIdx.x & 31;
  if (gid >= G) return;
  const float* f = fp + (long long)gid * INNER;
  float r[16];
#pragma unroll
  for (int j = 0; j < 16; ++j) r[j] = f[lane + 32 * j];
  float acc = 0.0f;
  for (int h = 0; h < 50; ++h) {
    const float* w = l1w + h * INNER;
    float p = 0.0f;
#pragma unroll
    for (int j = 0; j < 16; ++j) p += r[j] * w[lane + 32 * j];
#pragma unroll
    for (int o = 16; o > 0; o >>= 1) p += __shfl_xor(p, o, 32);
    acc += (p + l1b[h]) * l2w[h];
  }
  if (lane == 0) out[gid] = 1.0f / (1.0f + __expf(-(acc + l2b[0])));
}

// ---------------------------------------------------------------------------
// Host orchestration (graph-capture safe: only kernel launches on `stream`)
// ---------------------------------------------------------------------------
extern "C" void kernel_launch(void* const* d_in, const int* in_sizes, int n_in,
                              void* d_out, int out_size, void* d_ws, size_t ws_size,
                              hipStream_t stream) {
  const float* x        = (const float*)d_in[0];
  const float* edg_attr = (const float*)d_in[1];
  const float* Win_w    = (const float*)d_in[2];
  const float* Win_b    = (const float*)d_in[3];
  const float* Wout_w   = (const float*)d_in[4];
  const float* Wout_b   = (const float*)d_in[5];
  const float* l1w      = (const float*)d_in[6];
  const float* l1b      = (const float*)d_in[7];
  const float* l2w      = (const float*)d_in[8];
  const float* l2b      = (const float*)d_in[9];
  const int*   ei       = (const int*)d_in[10];
  const int*   batch    = (const int*)d_in[11];
  float*       out      = (float*)d_out;

  const int N = in_sizes[0] / XD;
  const int E = in_sizes[1] / ED;
  const int G = out_size;                 // reference output is [G, 1]

  // Workspace carve-up (all offsets stay 32B-aligned)
  char* ws = (char*)d_ws;
  float* x_buf = (float*)ws; ws += (size_t)N * XD * sizeof(float);
  float* aggx  = (float*)ws; ws += (size_t)N * XD * sizeof(float);
  float* agge  = (float*)ws; ws += (size_t)N * ED * sizeof(float);
  float* fp    = (float*)ws; ws += (size_t)G * INNER * sizeof(float);
  v16h*  woutP = (v16h*)ws;  ws += (size_t)2 * 1024 * sizeof(v16h);   // 64 KB
  v16h*  winP  = (v16h*)ws;  ws += (size_t)3 * 128 * sizeof(v16h);    // 12 KB

  const long long fp_n = (long long)G * INNER;
  const long long ax_n = (long long)N * XD;
  const long long ae_n = (long long)N * ED;
  const long long sx_n = (long long)E * XD;
  const long long se_n = (long long)E * ED;

  // One-time (per launch) weight packing + fp clear
  pack_wout_kernel<<<4, 256, 0, stream>>>(Wout_w, woutP);
  pack_win_kernel<<<1, 128, 0, stream>>>(Win_w, winP);
  zero_f32_kernel<<<(unsigned)((fp_n + 255) / 256), 256, 0, stream>>>(fp, fp_n);

  const int tiles = (N + 15) / 16;
  const float* xc = x;
  for (int depth = 0; depth <= 4; ++depth) {
    softmax_fp_kernel<<<tiles, 256, 0, stream>>>(xc, woutP, Wout_b, batch, fp, N);
    if (depth < 4) {
      // self-loop: agg_x starts as current x; agg_e starts at zero
      copy_f32_kernel<<<(unsigned)((ax_n + 255) / 256), 256, 0, stream>>>(aggx, xc, ax_n);
      zero_f32_kernel<<<(unsigned)((ae_n + 255) / 256), 256, 0, stream>>>(agge, ae_n);
      scatter_x_kernel<<<(unsigned)((sx_n + 255) / 256), 256, 0, stream>>>(xc, ei, aggx, E);
      scatter_e_kernel<<<(unsigned)((se_n + 255) / 256), 256, 0, stream>>>(edg_attr, ei, agge, E);
      // conv reads only aggx/agge, so writing x_buf (even when xc==x_buf) is safe
      conv_mm_kernel<<<tiles, 128, 0, stream>>>(aggx, agge, winP, Win_b, x_buf, N);
      xc = x_buf;
    }
  }

  mlp_kernel<<<(G + 7) / 8, 256, 0, stream>>>(fp, l1w, l1b, l2w, l2b, out, G);
}
